// Model_14534169330357
// MI455X (gfx1250) — matrix operand order repaired and checked by reading, not compiled, not run on hardware
//
#include <hip/hip_runtime.h>
#include <math.h>

#define BB   64
#define TT   512
#define HH   512
#define II   300
#define IPAD 320
#define NWG  32
#define NTHR 256

typedef __attribute__((ext_vector_type(16))) _Float16 v16h;
typedef __attribute__((ext_vector_type(8)))  _Float16 v8h;
typedef __attribute__((ext_vector_type(8)))  float    v8f;

union V16U { v16h v; v8h h[2]; };

__device__ __forceinline__ v16h pack16(v8h a, v8h b) {
    V16U u; u.h[0] = a; u.h[1] = b; return u.v;
}

__device__ __forceinline__ v8f wmma_f16(v16h a, v16h b, v8f c) {
    // D = A(16x32 f16) * B(32x16 f16) + C(16x16 f32)
    return __builtin_amdgcn_wmma_f32_16x16x32_f16(false, a, false, b, (short)0, c,
                                                  false, false);
}

__device__ __forceinline__ float sigm(float x) { return 1.0f / (1.0f + __expf(-x)); }

// ---------------- grid-wide two-phase barrier ----------------
__device__ __forceinline__ void grid_barrier(unsigned* cnt, unsigned* gen) {
    __syncthreads();
    if (threadIdx.x == 0) {
        __threadfence();
        unsigned g = __hip_atomic_load(gen, __ATOMIC_RELAXED, __HIP_MEMORY_SCOPE_AGENT);
        unsigned v = __hip_atomic_fetch_add(cnt, 1u, __ATOMIC_ACQ_REL, __HIP_MEMORY_SCOPE_AGENT);
        if (v == (unsigned)NWG - 1u) {
            __hip_atomic_store(cnt, 0u, __ATOMIC_RELAXED, __HIP_MEMORY_SCOPE_AGENT);
            __hip_atomic_fetch_add(gen, 1u, __ATOMIC_ACQ_REL, __HIP_MEMORY_SCOPE_AGENT);
        } else {
            while (__hip_atomic_load(gen, __ATOMIC_ACQUIRE, __HIP_MEMORY_SCOPE_AGENT) == g) {
                __builtin_amdgcn_s_sleep(1);
            }
        }
    }
    __syncthreads();
    __threadfence();
}

// ---------------- embedding gather + f16 convert (K padded to 320) ----------------
__global__ void embed_gather(const int* __restrict__ X, const float* __restrict__ emb,
                             _Float16* __restrict__ x0) {
    int row = blockIdx.x;                 // b*T + t
    int tok = X[row];
    const float* e = emb + (size_t)tok * II;
    _Float16* o = x0 + (size_t)row * IPAD;
    for (int k = threadIdx.x; k < IPAD; k += blockDim.x)
        o[k] = (k < II) ? (_Float16)e[k] : (_Float16)0.0f;
}

// ---------------- persistent 2-layer LSTM ----------------
// 32 WGs; WG wg owns hidden units [wg*16, wg*16+16) -> 64 gate cols (i,f,g,o x16).
// Per wave: one batch tile (mt), two adjacent gate tiles. The x-projection for
// step t+1 is computed BEFORE the inter-step grid barrier so only the h-GEMM
// (the true recurrence) sits on the serialized critical path.
__global__ void __launch_bounds__(NTHR, 1)
lstm2_persistent(const _Float16* __restrict__ x0,
                 _Float16* __restrict__ hs0,
                 _Float16* __restrict__ hbufs,   // [2][B][H] f16 ping-pong
                 _Float16* __restrict__ hfinal,  // [B][H] f16
                 const float* __restrict__ wih0, const float* __restrict__ whh0,
                 const float* __restrict__ bih0, const float* __restrict__ bhh0,
                 const float* __restrict__ wih1, const float* __restrict__ whh1,
                 const float* __restrict__ bih1, const float* __restrict__ bhh1,
                 unsigned* __restrict__ bar_cnt, unsigned* __restrict__ bar_gen) {
    extern __shared__ char smem[];
    _Float16* WIH  = (_Float16*)smem;                       // [64][512], B^T: [n][k]
    _Float16* WHH  = (_Float16*)(smem + 64 * 512 * 2);      // [64][512]
    float*    gatesL = (float*)(smem + 2 * 64 * 512 * 2);   // [64][68]
    float*    cL     = gatesL + 64 * 68;                    // [64][16]
    float*    biasL  = cL + 64 * 16;                        // [64]

    const int tid  = threadIdx.x;
    const int wave = tid >> 5;     // wave32: 8 waves
    const int lane = tid & 31;
    const int lo16 = lane & 15;
    const int hi   = lane >> 4;
    const int base = blockIdx.x * 16;

    const int mt  = wave >> 1;
    const int nt0 = (wave & 1) * 2;
    const int n0  = nt0 * 16 + lo16;

#pragma unroll 2   // clone per layer so Kx/rlen become compile-time constants
    for (int layer = 0; layer < 2; ++layer) {
        const float* wih = layer ? wih1 : wih0;
        const float* whh = layer ? whh1 : whh0;
        const float* bih = layer ? bih1 : bih0;
        const float* bhh = layer ? bhh1 : bhh0;
        const int rlen = layer ? HH : II;      // valid K in w_ih rows
        const int Kx   = layer ? HH : IPAD;    // padded K / row stride of xin
        const _Float16* xin = layer ? hs0 : x0;

        // x-projection for one timestep: a0/a1 += x_t @ Wih^T (two gate tiles)
        auto xproj = [&](int t, v8f& a0, v8f& a1) {
            const _Float16* ap = xin + ((size_t)(mt * 16 + lo16) * TT + t) * Kx + 16 * hi;
            const _Float16* bp = WIH + n0 * 512 + 16 * hi;   // nt0; nt0+1 at +16*512
            for (int kb = 0; kb < Kx; kb += 32) {
                v16h av  = pack16(*(const v8h*)(ap + kb), *(const v8h*)(ap + kb + 8));
                v16h bv0 = pack16(*(const v8h*)(bp + kb), *(const v8h*)(bp + kb + 8));
                v16h bv1 = pack16(*(const v8h*)(bp + 16 * 512 + kb),
                                  *(const v8h*)(bp + 16 * 512 + kb + 8));
                a0 = wmma_f16(av, bv0, a0);
                a1 = wmma_f16(av, bv1, a1);
            }
        };

        // --- load this WG's weight slice into LDS (f32 -> f16), B laid out [n][k] ---
        for (int idx = tid; idx < 64 * Kx; idx += NTHR) {
            int n = idx / Kx, k = idx - n * Kx;
            int c = (n >> 4) * HH + base + (n & 15);           // gate-major row of w_ih
            WIH[n * 512 + k] = (k < rlen) ? (_Float16)wih[(size_t)c * rlen + k]
                                          : (_Float16)0.0f;
        }
        for (int idx = tid; idx < 64 * HH; idx += NTHR) {
            int n = idx >> 9, k = idx & 511;
            int c = (n >> 4) * HH + base + (n & 15);
            WHH[n * 512 + k] = (_Float16)whh[(size_t)c * HH + k];
        }
        if (tid < 64) {
            int c = (tid >> 4) * HH + base + (tid & 15);
            biasL[tid] = bih[c] + bhh[c];
        }
        for (int idx = tid; idx < 64 * 16; idx += NTHR) cL[idx] = 0.0f;
        // zero h ping buffer 0 grid-wide (h_0 = 0)
        for (int i = tid + blockIdx.x * NTHR; i < BB * HH; i += NWG * NTHR)
            hbufs[i] = (_Float16)0.0f;
        __syncthreads();                        // WIH ready for this WG

        // prefetch x-projection for t=0 (overlaps other WGs' setup)
        v8f xa0 = {}, xa1 = {};
        xproj(0, xa0, xa1);
        grid_barrier(bar_cnt, bar_gen);

        for (int t = 0; t < TT; ++t) {
            const _Float16* hprev = hbufs + ((t & 1) ? BB * HH : 0);
            _Float16*       hnext = hbufs + ((t & 1) ? 0 : BB * HH);

            // recurrent projection on the critical path: gates = xa + h_{t-1} @ Whh^T
            v8f acc0 = xa0, acc1 = xa1;
            {
                const _Float16* hp = hprev + (size_t)(mt * 16 + lo16) * HH + 16 * hi;
                const _Float16* wp = WHH + n0 * 512 + 16 * hi;
                for (int kb = 0; kb < HH; kb += 32) {
                    v16h av  = pack16(*(const v8h*)(hp + kb), *(const v8h*)(hp + kb + 8));
                    v16h bv0 = pack16(*(const v8h*)(wp + kb), *(const v8h*)(wp + kb + 8));
                    v16h bv1 = pack16(*(const v8h*)(wp + 16 * 512 + kb),
                                      *(const v8h*)(wp + 16 * 512 + kb + 8));
                    acc0 = wmma_f16(av, bv0, acc0);
                    acc1 = wmma_f16(av, bv1, acc1);
                }
            }
            // C/D layout: lane holds (m = mt*16 + 8*hi + v, n = n0 [+16])
#pragma unroll
            for (int v = 0; v < 8; ++v) {
                gatesL[(mt * 16 + 8 * hi + v) * 68 + n0]      = acc0[v];
                gatesL[(mt * 16 + 8 * hi + v) * 68 + n0 + 16] = acc1[v];
            }
            __syncthreads();

            // elementwise LSTM cell for this 16-unit slice (f32 state)
#pragma unroll
            for (int u = 0; u < 4; ++u) {
                int id = u * NTHR + tid;           // 0..1023 = 64 rows x 16 cols
                int b = id >> 4, jj = id & 15;
                float gi = sigm(gatesL[b * 68 + jj]        + biasL[jj]);
                float gf = sigm(gatesL[b * 68 + 16 + jj]   + biasL[16 + jj]);
                float gg = tanhf(gatesL[b * 68 + 32 + jj]  + biasL[32 + jj]);
                float go = sigm(gatesL[b * 68 + 48 + jj]   + biasL[48 + jj]);
                float c = gf * cL[b * 16 + jj] + gi * gg;
                cL[b * 16 + jj] = c;
                float h = go * tanhf(c);
                _Float16 hv = (_Float16)h;
                hnext[(size_t)b * HH + base + jj] = hv;
                if (layer == 0)
                    hs0[((size_t)b * TT + t) * HH + base + jj] = hv;
                else if (t == TT - 1)
                    hfinal[(size_t)b * HH + base + jj] = hv;
            }

            // prefetch next step's x-projection BEFORE the barrier (off critical path);
            // grid-uniform guard keeps EXEC all-ones around WMMA
            xa0 = v8f{}; xa1 = v8f{};
            if (t + 1 < TT) xproj(t + 1, xa0, xa1);

            grid_barrier(bar_cnt, bar_gen);
        }
    }
}

// ---------------- final head: sigmoid(h2 @ w_out^T + b_out) ----------------
__global__ void logits_head(const _Float16* __restrict__ hfinal,
                            const float* __restrict__ w_out,
                            const float* __restrict__ b_out,
                            float* __restrict__ out) {
    int id = threadIdx.x;            // 128 threads: (b, k)
    if (id < BB * 2) {
        int b = id >> 1, k = id & 1;
        float s = b_out[k];
        const _Float16* hp = hfinal + (size_t)b * HH;
        const float* wp = w_out + (size_t)k * HH;
        for (int j = 0; j < HH; ++j) s += (float)hp[j] * wp[j];
        out[b * 2 + k] = 1.0f / (1.0f + __expf(-s));
    }
}

// ---------------- workspace layout ----------------
static const size_t OFF_X0  = 0;                                   // [B*T][320] f16
static const size_t SZ_X0   = (size_t)BB * TT * IPAD * 2;
static const size_t OFF_HS0 = OFF_X0 + SZ_X0;                      // [B*T][512] f16
static const size_t SZ_HS0  = (size_t)BB * TT * HH * 2;
static const size_t OFF_HB  = OFF_HS0 + SZ_HS0;                    // 2x[B][H] f16
static const size_t SZ_HB   = (size_t)2 * BB * HH * 2;
static const size_t OFF_HF  = OFF_HB + SZ_HB;                      // [B][H] f16
static const size_t SZ_HF   = (size_t)BB * HH * 2;
static const size_t OFF_BAR = OFF_HF + SZ_HF;                      // barrier counters
static const size_t SMEM_BYTES = 2 * 64 * 512 * 2 + 64 * 68 * 4 + 64 * 16 * 4 + 64 * 4;

extern "C" void kernel_launch(void* const* d_in, const int* in_sizes, int n_in,
                              void* d_out, int out_size, void* d_ws, size_t ws_size,
                              hipStream_t stream) {
    const int*   X    = (const int*)d_in[0];
    const float* emb  = (const float*)d_in[1];
    const float* wih0 = (const float*)d_in[2];
    const float* whh0 = (const float*)d_in[3];
    const float* bih0 = (const float*)d_in[4];
    const float* bhh0 = (const float*)d_in[5];
    const float* wih1 = (const float*)d_in[6];
    const float* whh1 = (const float*)d_in[7];
    const float* bih1 = (const float*)d_in[8];
    const float* bhh1 = (const float*)d_in[9];
    const float* wout = (const float*)d_in[10];
    const float* bout = (const float*)d_in[11];
    float* out = (float*)d_out;

    char* ws = (char*)d_ws;
    _Float16* x0     = (_Float16*)(ws + OFF_X0);
    _Float16* hs0    = (_Float16*)(ws + OFF_HS0);
    _Float16* hbufs  = (_Float16*)(ws + OFF_HB);
    _Float16* hfinal = (_Float16*)(ws + OFF_HF);
    unsigned* barp   = (unsigned*)(ws + OFF_BAR);

    // reset barrier counters every call (workspace is not re-poisoned between replays)
    hipMemsetAsync(barp, 0, 256, stream);

    embed_gather<<<BB * TT, 64, 0, stream>>>(X, emb, x0);

    lstm2_persistent<<<NWG, NTHR, SMEM_BYTES, stream>>>(
        x0, hs0, hbufs, hfinal,
        wih0, whh0, bih0, bhh0,
        wih1, whh1, bih1, bhh1,
        barp, barp + 1);

    logits_head<<<1, 128, 0, stream>>>(hfinal, wout, bout, out);
}
